// SlotAttention_10428180595019
// MI455X (gfx1250) — matrix operand order repaired and checked by reading, not compiled, not run on hardware
//
#include <hip/hip_runtime.h>

// ---------------------------------------------------------------------------
// SlotAttention for MI455X (gfx1250, wave32, WMMA bf16 16x16x32, f32 accum)
// + async global->LDS staging (ASYNCcnt) and global prefetch
// ---------------------------------------------------------------------------

#define N_SLOTS 16
#define NDIM    512
#define NSEQ    1024
#define NBATCH  64
#define SCALE_F 0.04419417382415922f   // 512^-0.5
#define EPS_F   1e-8f
#define LNEPS_F 1e-5f

typedef __attribute__((ext_vector_type(16))) __bf16 v16bf;
typedef __attribute__((ext_vector_type(8)))  float  v8f;

union BfFrag { v16bf v; unsigned int u[8]; unsigned short s[16]; };

__device__ __forceinline__ unsigned short f2bf(float f) {
    unsigned int u = __float_as_uint(f);
    unsigned int r = (u + 0x7FFFu + ((u >> 16) & 1u)) >> 16;   // RNE
    return (unsigned short)r;
}

__device__ __forceinline__ v8f wmma_bf16(v16bf a, v16bf b, v8f c) {
    return __builtin_amdgcn_wmma_f32_16x16x32_bf16(
        /*neg_a=*/false, a, /*neg_b=*/false, b,
        /*c_mod=*/(short)0, c, /*reuse_a=*/false, /*reuse_b=*/false);
}

// A fragment: 16x32 bf16 (MxK), row-major source with leading dim ld (elems).
// ISA 7.12.2: lane L holds row M=L%16; lane group g=L/16 selects K pattern:
//   elements 0..7  -> K = 8*g   .. 8*g+7
//   elements 8..15 -> K = 16+8*g .. 16+8*g+7
__device__ __forceinline__ v16bf load_a_frag(const unsigned short* base, int ld, int k0) {
    int lane = threadIdx.x & 31;
    int g = lane >> 4, m = lane & 15;
    const unsigned short* p = base + m * ld + k0;
    BfFrag f;
    uint4 a = *(const uint4*)(p + 8 * g);
    uint4 b = *(const uint4*)(p + 16 + 8 * g);
    f.u[0] = a.x; f.u[1] = a.y; f.u[2] = a.z; f.u[3] = a.w;
    f.u[4] = b.x; f.u[5] = b.y; f.u[6] = b.z; f.u[7] = b.w;
    return f.v;
}

// B fragment: 32x16 bf16 (KxN) taken from a TRANSPOSED operand stored [N][K]
// row-major (leading dim ld). Lane L: column N = L%16, K = 8*(L/16)+e (e<8), 16+8*(L/16)+(e-8) (e>=8).
// base must already point at row n0 of the [N][K] operand.
__device__ __forceinline__ v16bf load_b_frag(const unsigned short* base, int ld, int k0) {
    int lane = threadIdx.x & 31;
    int n = lane & 15, gk = lane >> 4;
    const unsigned short* p = base + n * ld + k0;
    BfFrag f;
    uint4 a = *(const uint4*)(p + 8 * gk);
    uint4 b = *(const uint4*)(p + 16 + 8 * gk);
    f.u[0] = a.x; f.u[1] = a.y; f.u[2] = a.z; f.u[3] = a.w;
    f.u[4] = b.x; f.u[5] = b.y; f.u[6] = b.z; f.u[7] = b.w;
    return f.v;
}

// Prefetch the leading cachelines of a 16-row x (ld elems) bf16 tile into the
// WGP cache hierarchy (lowers to global_prefetch_b8).
__device__ __forceinline__ void prefetch_tile(const unsigned short* base, int ld) {
    int lane = threadIdx.x & 31;
    const char* p = (const char*)(base + (size_t)(lane & 15) * ld) + (lane >> 4) * 128;
    __builtin_prefetch(p, 0, 1);
}

// ---------------------------------------------------------------------------
// Kernel: transpose + convert f32 weight [K][N] -> bf16 [N][K]
// ---------------------------------------------------------------------------
__global__ void tconv_kernel(const float* __restrict__ in, unsigned short* __restrict__ out,
                             int Kd, int Nd) {
    int idx = blockIdx.x * 256 + threadIdx.x;
    if (idx >= Kd * Nd) return;
    int n = idx / Kd, k = idx - n * Kd;
    out[idx] = f2bf(in[(size_t)k * Nd + n]);
}

// ---------------------------------------------------------------------------
// Kernel: LayerNorm(X) -> bf16, one row (D=512) per wave32
// ---------------------------------------------------------------------------
__global__ __launch_bounds__(256) void ln_x_kernel(const float* __restrict__ X,
                                                   const float* __restrict__ g,
                                                   const float* __restrict__ bta,
                                                   unsigned short* __restrict__ Xn,
                                                   int nrows) {
    int wave = threadIdx.x >> 5, lane = threadIdx.x & 31;
    int row = blockIdx.x * 8 + wave;
    if (row >= nrows) return;
    const float* src = X + (size_t)row * NDIM;
    float x[16], sum = 0.f, sq = 0.f;
    #pragma unroll
    for (int j = 0; j < 16; ++j) {
        x[j] = src[lane + 32 * j];
        sum += x[j]; sq += x[j] * x[j];
    }
    #pragma unroll
    for (int off = 16; off > 0; off >>= 1) {
        sum += __shfl_xor(sum, off, 32);
        sq  += __shfl_xor(sq,  off, 32);
    }
    float mean = sum * (1.f / NDIM);
    float var  = sq  * (1.f / NDIM) - mean * mean;
    float inv  = rsqrtf(var + LNEPS_F);
    unsigned short* dst = Xn + (size_t)row * NDIM;
    #pragma unroll
    for (int j = 0; j < 16; ++j) {
        int c = lane + 32 * j;
        dst[c] = f2bf((x[j] - mean) * inv * g[c] + bta[c]);
    }
}

// ---------------------------------------------------------------------------
// Kernel: kv = LN(X) @ W_in + b_in   (bf16 WMMA, f32 accum)
// grid = (rowBlocks=1024, colBlocks=16); block = 256 (8 waves)
// Each block: 64 rows x 64 cols. k -> Kbuf[B][S][D], v -> VT[B][D][S] (transposed)
// A-tile staged into LDS with GLOBAL_LOAD_ASYNC_TO_LDS_B128 (ASYNCcnt).
// ---------------------------------------------------------------------------
__global__ __launch_bounds__(256) void kv_gemm_kernel(
        const unsigned short* __restrict__ Xn,
        const unsigned short* __restrict__ WinT,   // [1024][512] bf16
        const float* __restrict__ b_in,
        unsigned short* __restrict__ Kbuf,
        unsigned short* __restrict__ VT) {
    extern __shared__ char smem[];
    unsigned short* sA = (unsigned short*)smem;    // 64 x 512 bf16 = 64 KB, LDS offset 0
    int rb = blockIdx.x, cb = blockIdx.y;
    int tid = threadIdx.x, wave = tid >> 5, lane = tid & 31;

    // async copy of 64 contiguous LN'ed rows (64 KB) direct to LDS.
    // Dynamic LDS starts at offset 0 in this kernel (no static LDS in TU),
    // so the LDS destination address is simply the byte offset into smem.
    {
        const unsigned short* srcBase = Xn + (size_t)rb * 64 * NDIM;
        for (int i = tid; i < 64 * NDIM / 8; i += 256) {      // 4096 x 16B, 16 per thread
            unsigned int off16 = (unsigned int)i * 16u;       // LDS byte addr == global byte off
            asm volatile("global_load_async_to_lds_b128 %0, %1, %2"
                         :
                         : "v"(off16), "v"(off16), "s"(srcBase)
                         : "memory");
        }
        asm volatile("s_wait_asynccnt 0x0" ::: "memory");
    }
    __syncthreads();

    for (int j = 0; j < 2; ++j) {
        int t = wave + 8 * j;              // 0..15 -> 4x4 tile grid
        int tm = t >> 2, tn = t & 3;
        int m0 = tm * 16;
        int n0 = cb * 64 + tn * 16;        // global output column
        v8f acc = {};
        for (int k0 = 0; k0 < NDIM; k0 += 32) {
            acc = wmma_bf16(load_a_frag(sA + m0 * NDIM, NDIM, k0),
                            load_b_frag(WinT + (size_t)n0 * NDIM, NDIM, k0), acc);
        }
        int g = lane >> 4, n = lane & 15;
        #pragma unroll
        for (int i = 0; i < 8; ++i) {
            int m = m0 + i + 8 * g;
            size_t rg = (size_t)rb * 64 + m;
            int b = (int)(rg >> 10);
            int s = (int)(rg & 1023);
            int nn = n0 + n;
            unsigned short h = f2bf(acc[i] + b_in[nn]);
            if (nn < NDIM)
                Kbuf[((size_t)b * NSEQ + s) * NDIM + nn] = h;
            else
                VT[((size_t)b * NDIM + (nn - NDIM)) * NSEQ + s] = h;
        }
    }
}

// ---------------------------------------------------------------------------
// Fused per-batch iterative slot-attention kernel (1 block = 1 batch, 8 waves)
// ---------------------------------------------------------------------------
__device__ __forceinline__ void ln_rows16(const float* src, const float* g, const float* bta,
                                          unsigned short* dst, int ldd, int wave, int lane) {
    for (int r = wave; r < N_SLOTS; r += 8) {
        float x[16], sum = 0.f, sq = 0.f;
        #pragma unroll
        for (int j = 0; j < 16; ++j) {
            x[j] = src[r * NDIM + lane + 32 * j];
            sum += x[j]; sq += x[j] * x[j];
        }
        #pragma unroll
        for (int off = 16; off > 0; off >>= 1) {
            sum += __shfl_xor(sum, off, 32);
            sq  += __shfl_xor(sq,  off, 32);
        }
        float mean = sum * (1.f / NDIM);
        float var  = sq  * (1.f / NDIM) - mean * mean;
        float inv  = rsqrtf(var + LNEPS_F);
        #pragma unroll
        for (int j = 0; j < 16; ++j) {
            int c = lane + 32 * j;
            dst[r * ldd + c] = f2bf((x[j] - mean) * inv * g[c] + bta[c]);
        }
    }
}

__global__ __launch_bounds__(256) void slot_iter_kernel(
        const float* __restrict__ prev_slots,
        const unsigned short* __restrict__ Kbuf,   // [B][S][D] bf16
        const unsigned short* __restrict__ VT,     // [B][D][S] bf16
        const unsigned short* __restrict__ WqT,  const float* __restrict__ b_q,
        const unsigned short* __restrict__ WihT, const float* __restrict__ b_ih,
        const unsigned short* __restrict__ WhhT, const float* __restrict__ b_hh,
        const unsigned short* __restrict__ W1T,  const float* __restrict__ b1,
        const unsigned short* __restrict__ W2T,  const float* __restrict__ b2,
        const float* __restrict__ sn_g,  const float* __restrict__ sn_b,
        const float* __restrict__ sn2_g, const float* __restrict__ sn2_b,
        float* __restrict__ out) {
    extern __shared__ char smem[];
    float*          sSlots = (float*)smem;                          // 16x512 f32  (32 KB)
    float*          sPrev  = (float*)(smem + 32 * 1024);            // 16x512 f32  (32 KB)
    float*          sC     = (float*)(smem + 64 * 1024);            // 16x1024 f32 (64 KB)
    unsigned short* sA     = (unsigned short*)(smem + 128 * 1024);  // 16x1024 bf16(32 KB)
    unsigned short* sU     = (unsigned short*)(smem + 160 * 1024);  // 16x512 bf16 (16 KB)
    float*          sRed   = (float*)(smem + 176 * 1024);           // 16 f32

    int b = blockIdx.x;
    int tid = threadIdx.x, wave = tid >> 5, lane = tid & 31;
    const unsigned short* Kb = Kbuf + (size_t)b * NSEQ * NDIM;
    const unsigned short* Vb = VT   + (size_t)b * NDIM * NSEQ;

    for (int i = tid; i < N_SLOTS * NDIM; i += 256)
        sSlots[i] = prev_slots[(size_t)b * N_SLOTS * NDIM + i];
    __syncthreads();

    for (int iter = 0; iter < 3; ++iter) {
        // slots_prev = slots
        for (int i = tid; i < N_SLOTS * NDIM; i += 256) sPrev[i] = sSlots[i];
        __syncthreads();

        // --- q = LN(slots; sn) @ WqT + b_q  -> bf16 in sU ---
        ln_rows16(sSlots, sn_g, sn_b, sA, 1024, wave, lane);
        __syncthreads();
        for (int j = 0; j < 4; ++j) {
            int n0 = (wave + 8 * j) * 16;                  // N = 512
            v8f acc = {};
            for (int k0 = 0; k0 < NDIM; k0 += 32)
                acc = wmma_bf16(load_a_frag(sA, 1024, k0),
                                load_b_frag(WqT + (size_t)n0 * NDIM, NDIM, k0), acc);
            int g = lane >> 4, n = lane & 15;
            #pragma unroll
            for (int i = 0; i < 8; ++i)
                sU[(i + 8 * g) * NDIM + n0 + n] = f2bf(acc[i] + b_q[n0 + n]);
        }
        __syncthreads();

        // --- logits = SCALE * q @ k^T  -> f32 [16,1024] in sC ---
        for (int j = 0; j < 8; ++j) {
            int n0 = (wave + 8 * j) * 16;                  // N = 1024 (= S)
            if (j + 1 < 8)                                 // prefetch next tile of k
                prefetch_tile(Kb + (size_t)(wave + 8 * (j + 1)) * 16 * NDIM, NDIM);
            v8f acc = {};
            for (int k0 = 0; k0 < NDIM; k0 += 32)
                acc = wmma_bf16(load_a_frag(sU, NDIM, k0),
                                load_b_frag(Kb + (size_t)n0 * NDIM, NDIM, k0), acc);
            int g = lane >> 4, n = lane & 15;
            #pragma unroll
            for (int i = 0; i < 8; ++i)
                sC[(i + 8 * g) * NSEQ + n0 + n] = acc[i] * SCALE_F;
        }
        __syncthreads();

        // --- softmax over slot axis (16 per column), +EPS ---
        for (int s = tid; s < NSEQ; s += 256) {
            float mx = -1e30f;
            #pragma unroll
            for (int i = 0; i < N_SLOTS; ++i) mx = fmaxf(mx, sC[i * NSEQ + s]);
            float e[N_SLOTS], sum = 0.f;
            #pragma unroll
            for (int i = 0; i < N_SLOTS; ++i) { e[i] = __expf(sC[i * NSEQ + s] - mx); sum += e[i]; }
            float inv = 1.f / sum;
            #pragma unroll
            for (int i = 0; i < N_SLOTS; ++i) sC[i * NSEQ + s] = e[i] * inv + EPS_F;
        }
        __syncthreads();
        // row sums over S (L1 renorm denominator)
        for (int r = wave; r < N_SLOTS; r += 8) {
            float s = 0.f;
            for (int c = lane; c < NSEQ; c += 32) s += sC[r * NSEQ + c];
            #pragma unroll
            for (int off = 16; off > 0; off >>= 1) s += __shfl_xor(s, off, 32);
            if (lane == 0) sRed[r] = s;
        }
        __syncthreads();
        // attn bf16 (renorm and fold /N_SLOTS) -> sA
        for (int i = tid; i < N_SLOTS * NSEQ; i += 256) {
            int r = i >> 10;
            sA[i] = f2bf(sC[i] / (sRed[r] * (float)N_SLOTS));
        }
        __syncthreads();

        // --- updates = attn @ v  -> bf16 in sU ---
        for (int j = 0; j < 4; ++j) {
            int n0 = (wave + 8 * j) * 16;                  // N = 512 (= D)
            if (j + 1 < 4)                                 // prefetch next tile of v^T
                prefetch_tile(Vb + (size_t)(wave + 8 * (j + 1)) * 16 * NSEQ, NSEQ);
            v8f acc = {};
            for (int k0 = 0; k0 < NSEQ; k0 += 32)
                acc = wmma_bf16(load_a_frag(sA, NSEQ, k0),
                                load_b_frag(Vb + (size_t)n0 * NSEQ, NSEQ, k0), acc);
            int g = lane >> 4, n = lane & 15;
            #pragma unroll
            for (int i = 0; i < 8; ++i)
                sU[(i + 8 * g) * NDIM + n0 + n] = f2bf(acc[i]);
        }
        __syncthreads();
        // prev slots -> bf16 A operand in sA (ld 1024, cols 0..511)
        for (int i = tid; i < N_SLOTS * NDIM; i += 256) {
            int r = i >> 9, c = i & 511;
            sA[r * 1024 + c] = f2bf(sPrev[i]);
        }
        __syncthreads();

        // --- GRU pass A: r,z = sigmoid(gi+gh) for gate cols [0,1024) -> sC ---
        for (int j = 0; j < 8; ++j) {
            int n0 = (wave + 8 * j) * 16;                  // 0..1023
            v8f acc = {};
            for (int k0 = 0; k0 < NDIM; k0 += 32) {
                acc = wmma_bf16(load_a_frag(sU, NDIM, k0),
                                load_b_frag(WihT + (size_t)n0 * NDIM, NDIM, k0), acc);
                acc = wmma_bf16(load_a_frag(sA, 1024, k0),
                                load_b_frag(WhhT + (size_t)n0 * NDIM, NDIM, k0), acc);
            }
            int g = lane >> 4, n = lane & 15;
            #pragma unroll
            for (int i = 0; i < 8; ++i) {
                int m = i + 8 * g, nn = n0 + n;
                float v = acc[i] + b_ih[nn] + b_hh[nn];
                sC[m * 1024 + nn] = 1.f / (1.f + __expf(-v));
            }
        }
        __syncthreads();

        // --- GRU pass B: n-gate (cols [1024,1536)) + combine -> new slots ---
        for (int j = 0; j < 4; ++j) {
            int t = 64 + wave + 8 * j;
            int n0 = t * 16;                               // 1024..1535
            v8f gi = {}, gh = {};
            for (int k0 = 0; k0 < NDIM; k0 += 32) {
                gi = wmma_bf16(load_a_frag(sU, NDIM, k0),
                               load_b_frag(WihT + (size_t)n0 * NDIM, NDIM, k0), gi);
                gh = wmma_bf16(load_a_frag(sA, 1024, k0),
                               load_b_frag(WhhT + (size_t)n0 * NDIM, NDIM, k0), gh);
            }
            int g = lane >> 4, n = lane & 15;
            #pragma unroll
            for (int i = 0; i < 8; ++i) {
                int m = i + 8 * g, nn = n0 + n, nl = nn - 1024;
                float r  = sC[m * 1024 + nl];
                float z  = sC[m * 1024 + 512 + nl];
                float nv = tanhf((gi[i] + b_ih[nn]) + r * (gh[i] + b_hh[nn]));
                float pv = sPrev[m * NDIM + nl];
                sSlots[m * NDIM + nl] = (1.f - z) * nv + z * pv;
            }
        }
        __syncthreads();

        // --- residual MLP: slots += relu(LN(slots; sn2) @ W1 + b1) @ W2 + b2 ---
        ln_rows16(sSlots, sn2_g, sn2_b, sA, 1024, wave, lane);
        __syncthreads();
        for (int j = 0; j < 8; ++j) {
            int n0 = (wave + 8 * j) * 16;                  // N = 1024
            v8f acc = {};
            for (int k0 = 0; k0 < NDIM; k0 += 32)
                acc = wmma_bf16(load_a_frag(sA, 1024, k0),
                                load_b_frag(W1T + (size_t)n0 * NDIM, NDIM, k0), acc);
            int g = lane >> 4, n = lane & 15;
            #pragma unroll
            for (int i = 0; i < 8; ++i)
                sC[(i + 8 * g) * 1024 + n0 + n] = acc[i] + b1[n0 + n];
        }
        __syncthreads();
        for (int i = tid; i < N_SLOTS * 1024; i += 256)
            sA[i] = f2bf(fmaxf(sC[i], 0.f));
        __syncthreads();
        for (int j = 0; j < 4; ++j) {
            int n0 = (wave + 8 * j) * 16;                  // N = 512
            v8f acc = {};
            for (int k0 = 0; k0 < 1024; k0 += 32)
                acc = wmma_bf16(load_a_frag(sA, 1024, k0),
                                load_b_frag(W2T + (size_t)n0 * 1024, 1024, k0), acc);
            int g = lane >> 4, n = lane & 15;
            #pragma unroll
            for (int i = 0; i < 8; ++i) {
                int m = i + 8 * g, nn = n0 + n;
                sSlots[m * NDIM + nn] += acc[i] + b2[nn];
            }
        }
        __syncthreads();
    }

    for (int i = tid; i < N_SLOTS * NDIM; i += 256)
        out[(size_t)b * N_SLOTS * NDIM + i] = sSlots[i];
}

// ---------------------------------------------------------------------------
// Host launcher
// ---------------------------------------------------------------------------
extern "C" void kernel_launch(void* const* d_in, const int* in_sizes, int n_in,
                              void* d_out, int out_size, void* d_ws, size_t ws_size,
                              hipStream_t stream) {
    (void)in_sizes; (void)n_in; (void)out_size; (void)ws_size;

    const float* X     = (const float*)d_in[0];
    const float* pslot = (const float*)d_in[1];
    const float* in_g  = (const float*)d_in[2];
    const float* in_b  = (const float*)d_in[3];
    const float* W_in  = (const float*)d_in[4];
    const float* b_in  = (const float*)d_in[5];
    const float* sn_g  = (const float*)d_in[6];
    const float* sn_b  = (const float*)d_in[7];
    const float* W_q   = (const float*)d_in[8];
    const float* b_q   = (const float*)d_in[9];
    const float* sn2_g = (const float*)d_in[10];
    const float* sn2_b = (const float*)d_in[11];
    const float* W1    = (const float*)d_in[12];
    const float* b1    = (const float*)d_in[13];
    const float* W2    = (const float*)d_in[14];
    const float* b2    = (const float*)d_in[15];
    const float* W_ih  = (const float*)d_in[16];
    const float* b_ih  = (const float*)d_in[17];
    const float* W_hh  = (const float*)d_in[18];
    const float* b_hh  = (const float*)d_in[19];

    char* ws = (char*)d_ws;
    size_t off = 0;
    auto carve = [&](size_t bytes) {
        char* p = ws + off;
        off += (bytes + 255) & ~(size_t)255;
        return p;
    };
    unsigned short* Xn   = (unsigned short*)carve((size_t)NBATCH * NSEQ * NDIM * 2);
    unsigned short* Kbuf = (unsigned short*)carve((size_t)NBATCH * NSEQ * NDIM * 2);
    unsigned short* VT   = (unsigned short*)carve((size_t)NBATCH * NDIM * NSEQ * 2);
    unsigned short* WinT = (unsigned short*)carve((size_t)1024 * 512 * 2);
    unsigned short* WqT  = (unsigned short*)carve((size_t)512 * 512 * 2);
    unsigned short* W1T  = (unsigned short*)carve((size_t)1024 * 512 * 2);
    unsigned short* W2T  = (unsigned short*)carve((size_t)512 * 1024 * 2);
    unsigned short* WihT = (unsigned short*)carve((size_t)1536 * 512 * 2);
    unsigned short* WhhT = (unsigned short*)carve((size_t)1536 * 512 * 2);

    // weight transpose+convert to bf16 [N][K]
    tconv_kernel<<<(512 * 1024 + 255) / 256, 256, 0, stream>>>(W_in, WinT, 512, 1024);
    tconv_kernel<<<(512 * 512  + 255) / 256, 256, 0, stream>>>(W_q,  WqT,  512, 512);
    tconv_kernel<<<(512 * 1024 + 255) / 256, 256, 0, stream>>>(W1,   W1T,  512, 1024);
    tconv_kernel<<<(1024 * 512 + 255) / 256, 256, 0, stream>>>(W2,   W2T,  1024, 512);
    tconv_kernel<<<(512 * 1536 + 255) / 256, 256, 0, stream>>>(W_ih, WihT, 512, 1536);
    tconv_kernel<<<(512 * 1536 + 255) / 256, 256, 0, stream>>>(W_hh, WhhT, 512, 1536);

    // LN(X) -> bf16
    ln_x_kernel<<<(NBATCH * NSEQ) / 8, 256, 0, stream>>>(X, in_g, in_b, Xn, NBATCH * NSEQ);

    // kv projection (bf16 WMMA), k row-major, v transposed
    dim3 gkv(1024, 16);
    kv_gemm_kernel<<<gkv, 256, 64 * 1024, stream>>>(Xn, WinT, b_in, Kbuf, VT);

    // fused iterative slot attention, one block per batch
    slot_iter_kernel<<<NBATCH, 256, 176 * 1024 + 256, stream>>>(
        pslot, Kbuf, VT, WqT, b_q, WihT, b_ih, WhhT, b_hh,
        W1T, b1, W2T, b2, sn_g, sn_b, sn2_g, sn2_b, (float*)d_out);
}
